// BaseDepthNet_28630251995154
// MI455X (gfx1250) — matrix operand order repaired and checked by reading, not compiled, not run on hardware
//
#include <hip/hip_runtime.h>
#include <hip/hip_bf16.h>
#include <math.h>

typedef __attribute__((ext_vector_type(16))) _Float16 v16h;
typedef __attribute__((ext_vector_type(8)))  _Float16 v8h;
typedef __attribute__((ext_vector_type(8)))  float    v8f;
typedef __attribute__((ext_vector_type(4)))  float    v4f;

#define NCAM  6
#define CIN   512
#define MID   256
#define DBINS 112
#define H_    32
#define W_    88
#define HW    2816          // 32*88
#define NPIX  (NCAM*HW)     // 16896
#define VXD   128
#define VYD   128
#define VZD   16

union FragH { v16h v; v8h h[2]; };

// ---------------- weight transform: w[OC][C][ntaps] f32 -> o[OC][ntaps][C] f16
__global__ void k_wt(const float* __restrict__ w, _Float16* __restrict__ o,
                     int OC, int C, int ntaps) {
  int i = blockIdx.x * blockDim.x + threadIdx.x;
  int total = OC * C * ntaps;
  if (i >= total) return;
  int oc = i / (C * ntaps);
  int r  = i % (C * ntaps);
  int ci = r / ntaps;
  int t  = r % ntaps;
  o[((size_t)oc * ntaps + t) * C + ci] = (_Float16)w[i];
}

// ---------------- img transpose: in (cam, ci, p) f32 -> out (cam, p, ci) f16
__global__ void k_imgT(const float* __restrict__ in, _Float16* __restrict__ out) {
  int i = blockIdx.x * blockDim.x + threadIdx.x;
  if (i >= NCAM * CIN * HW) return;
  int cam = i / (CIN * HW);
  int r   = i % (CIN * HW);
  int ci  = r / HW;
  int p   = r % HW;
  out[((size_t)cam * HW + p) * CIN + ci] = (_Float16)in[i];
}

// ---------------- SE path: per-(cam,channel) sigmoid gate
__global__ void k_se(const float* __restrict__ intr,
                     const float* __restrict__ w1, const float* __restrict__ b1,
                     const float* __restrict__ w2, const float* __restrict__ b2,
                     const float* __restrict__ rw, const float* __restrict__ rb,
                     const float* __restrict__ ew, const float* __restrict__ eb,
                     float* __restrict__ gate) {
  __shared__ float h1[MID], h2[MID], tb[MID];
  int cam = blockIdx.x, j = threadIdx.x;
  float f0 = intr[cam * 16 + 0], f1 = intr[cam * 16 + 5];
  float i0 = 1.0f / f0, i1 = 1.0f / f1;
  float sp = 1000.0f * sqrtf(i0 * i0 + i1 * i1);
  h1[j] = fmaxf(sp * w1[j] + b1[j], 0.0f);
  __syncthreads();
  float a = b2[j];
  for (int k = 0; k < MID; ++k) a += h1[k] * w2[k * MID + j];
  h2[j] = a;
  __syncthreads();
  a = rb[j];
  for (int k = 0; k < MID; ++k) a += rw[j * MID + k] * h2[k];
  tb[j] = fmaxf(a, 0.0f);
  __syncthreads();
  a = eb[j];
  for (int k = 0; k < MID; ++k) a += ew[j * MID + k] * tb[k];
  gate[cam * MID + j] = 1.0f / (1.0f + expf(-a));
}

// ---------------- implicit-GEMM 3x3 conv: 2 OC-tiles x 4 pixel-tiles per wave
// in : [cam][pix][C] f16 channels-last ; wt : [oc][NTAPS][C] f16
// MODE: 0 relu(s*x+b)*gate | 1 relu(s*x+b) | 2 relu(prev + s*x+b)   -> f16 out
template<int C, int NTAPS, int MODE>
__global__ __launch_bounds__(32)
void k_conv(const _Float16* __restrict__ in, const _Float16* __restrict__ wt,
            const float* __restrict__ sc, const float* __restrict__ bi,
            const float* __restrict__ gate, const _Float16* prev,
            _Float16* out_h) {
  const int lane   = threadIdx.x;
  const int cam    = blockIdx.z;
  const int octile = blockIdx.y * 32;   // two 16-wide OC tiles
  const int ptile  = blockIdx.x * 64;   // four 16-wide pixel tiles
  const int hiA    = (lane >= 16) ? 8 : 0;   // A frag: K = hiA + {0..7,16..23}
  const int hiB    = (lane >= 16) ? 8 : 0;   // B frag: K = hiB + {0..7,16..23}
  const int m      = lane & 15;

  int p[4], py[4], px[4];
#pragma unroll
  for (int t = 0; t < 4; ++t) {
    p[t]  = ptile + t * 16 + m;
    py[t] = p[t] / W_;
    px[t] = p[t] % W_;
  }

  const _Float16* aw0 = wt + ((size_t)(octile + m)      * NTAPS) * C + hiA;
  const _Float16* aw1 = wt + ((size_t)(octile + 16 + m) * NTAPS) * C + hiA;
  const _Float16* inc = in + (size_t)cam * HW * C + hiB;

  v8f acc[2][4];
#pragma unroll
  for (int o = 0; o < 2; ++o)
#pragma unroll
    for (int t = 0; t < 4; ++t) acc[o][t] = (v8f){};

  const v8h zero8 = {};

#pragma unroll 1
  for (int tap = 0; tap < NTAPS; ++tap) {
    const int dy = (NTAPS == 9) ? (tap / 3 - 1) : 0;
    const int dx = (NTAPS == 9) ? (tap % 3 - 1) : 0;
    const _Float16* bp[4];
    bool bv[4];
#pragma unroll
    for (int t = 0; t < 4; ++t) {
      int yy = py[t] + dy, xx = px[t] + dx;
      bv[t] = (yy >= 0) && (yy < H_) && (xx >= 0) && (xx < W_);
      bp[t] = inc + (size_t)(yy * W_ + xx) * C;
    }
    const _Float16* a0 = aw0 + (size_t)tap * C;
    const _Float16* a1 = aw1 + (size_t)tap * C;
#pragma unroll 2
    for (int cb = 0; cb < C; cb += 32) {
      FragH fa0, fa1, fb[4];
      fa0.h[0] = *(const v8h*)(a0 + cb);
      fa0.h[1] = *(const v8h*)(a0 + cb + 16);
      fa1.h[0] = *(const v8h*)(a1 + cb);
      fa1.h[1] = *(const v8h*)(a1 + cb + 16);
#pragma unroll
      for (int t = 0; t < 4; ++t) {
        fb[t].h[0] = bv[t] ? *(const v8h*)(bp[t] + cb)     : zero8;
        fb[t].h[1] = bv[t] ? *(const v8h*)(bp[t] + cb + 16) : zero8;
      }
#pragma unroll
      for (int t = 0; t < 4; ++t) {
        acc[0][t] = __builtin_amdgcn_wmma_f32_16x16x32_f16(false, fa0.v, false, fb[t].v,
                                                           (short)0, acc[0][t], false, false);
        acc[1][t] = __builtin_amdgcn_wmma_f32_16x16x32_f16(false, fa1.v, false, fb[t].v,
                                                           (short)0, acc[1][t], false, false);
      }
    }
  }

#pragma unroll
  for (int o = 0; o < 2; ++o) {
    const int ocb = octile + o * 16 + hiA;  // C/D frag: lane holds M = ocb+v
#pragma unroll
    for (int t = 0; t < 4; ++t) {
      size_t ob = ((size_t)cam * HW + p[t]) * MID + ocb;
      v8h pv = (MODE == 2) ? *(const v8h*)(prev + ob) : zero8;
      v8h r;
#pragma unroll
      for (int v = 0; v < 8; ++v) {
        int oc = ocb + v;
        float val = acc[o][t][v] * sc[oc] + bi[oc];
        if (MODE == 0)      val = fmaxf(val, 0.0f) * gate[cam * MID + oc];
        else if (MODE == 1) val = fmaxf(val, 0.0f);
        else                val = fmaxf(val + (float)pv[v], 0.0f);
        r[v] = (_Float16)val;
      }
      *(v8h*)(out_h + ob) = r;
    }
  }
}

// ---------------- depth head: 1x1 conv 256->112, +bias, f32 channels-last out
__global__ __launch_bounds__(32)
void k_head(const _Float16* __restrict__ in, const _Float16* __restrict__ wt,
            const float* __restrict__ bi, float* __restrict__ out_f) {
  const int C = MID;
  const int lane   = threadIdx.x;
  const int cam    = blockIdx.z;
  const int octile = blockIdx.y * 16;
  const int ptile  = blockIdx.x * 32;
  const int hiA    = (lane >= 16) ? 8 : 0;
  const int hiB    = (lane >= 16) ? 8 : 0;
  const int m      = lane & 15;

  const int p0 = ptile + m, p1 = ptile + 16 + m;
  const _Float16* ap  = wt + (size_t)(octile + m) * C + hiA;
  const _Float16* bp0 = in + ((size_t)cam * HW + p0) * C + hiB;
  const _Float16* bp1 = in + ((size_t)cam * HW + p1) * C + hiB;

  v8f acc0 = {}; v8f acc1 = {};
#pragma unroll 2
  for (int cb = 0; cb < C; cb += 32) {
    FragH fa, fb0, fb1;
    fa.h[0]  = *(const v8h*)(ap + cb);
    fa.h[1]  = *(const v8h*)(ap + cb + 16);
    fb0.h[0] = *(const v8h*)(bp0 + cb);
    fb0.h[1] = *(const v8h*)(bp0 + cb + 16);
    fb1.h[0] = *(const v8h*)(bp1 + cb);
    fb1.h[1] = *(const v8h*)(bp1 + cb + 16);
    acc0 = __builtin_amdgcn_wmma_f32_16x16x32_f16(false, fa.v, false, fb0.v,
                                                  (short)0, acc0, false, false);
    acc1 = __builtin_amdgcn_wmma_f32_16x16x32_f16(false, fa.v, false, fb1.v,
                                                  (short)0, acc1, false, false);
  }

  const int ocb = octile + hiA;
#pragma unroll
  for (int t = 0; t < 2; ++t) {
    v8f acc = t ? acc1 : acc0;
    int p = t ? p1 : p0;
    size_t ob = ((size_t)cam * HW + p) * DBINS + ocb;
    v4f r0, r1;
#pragma unroll
    for (int v = 0; v < 4; ++v) {
      r0[v] = acc[v]     + bi[ocb + v];
      r1[v] = acc[v + 4] + bi[ocb + v + 4];
    }
    *(v4f*)(out_f + ob)     = r0;
    *(v4f*)(out_f + ob + 4) = r1;
  }
}

// ---------------- softmax over 112 bins; output transposed to [cam][bin][pix]
__global__ void k_softmax(const float* __restrict__ depth, float* __restrict__ dpn) {
  int i = blockIdx.x * blockDim.x + threadIdx.x;
  if (i >= NPIX) return;
  const float* d = depth + (size_t)i * DBINS;
  float mx = -1e30f;
  for (int b = 0; b < DBINS; ++b) mx = fmaxf(mx, d[b]);
  float s = 0.0f;
  for (int b = 0; b < DBINS; ++b) s += expf(d[b] - mx);
  float inv = 1.0f / s;
  int cam = i / HW, p = i % HW;
  for (int b = 0; b < DBINS; ++b)
    dpn[((size_t)cam * DBINS + b) * HW + p] = expf(d[b] - mx) * inv;
}

// ---------------- trilinear 3D grid-sample + camera sum + mask normalize
__global__ void k_sample(const float* __restrict__ grid, const float* __restrict__ dpn,
                         float* __restrict__ out) {
  int i = blockIdx.x * blockDim.x + threadIdx.x;
  if (i >= VXD * VYD * VZD) return;
  int vz = i % VZD, vy = (i / VZD) % VYD, vx = i / (VZD * VYD);
  float agg = 0.0f, msum = 0.0f;
  for (int cam = 0; cam < NCAM; ++cam) {
    const float* g = grid + ((((size_t)cam * VXD + vx) * VYD + vy) * VZD + vz) * 3;
    float ix = ((g[0] + 1.0f) * W_    - 1.0f) * 0.5f;
    float iy = ((g[1] + 1.0f) * H_    - 1.0f) * 0.5f;
    float iz = ((g[2] + 1.0f) * DBINS - 1.0f) * 0.5f;
    float fx0 = floorf(ix), fy0 = floorf(iy), fz0 = floorf(iz);
    float fx = ix - fx0, fy = iy - fy0, fz = iz - fz0;
    int ix0 = (int)fx0, iy0 = (int)fy0, iz0 = (int)fz0;
#pragma unroll
    for (int dz = 0; dz < 2; ++dz)
#pragma unroll
      for (int dy = 0; dy < 2; ++dy)
#pragma unroll
        for (int dx = 0; dx < 2; ++dx) {
          int xi = ix0 + dx, yi = iy0 + dy, zi = iz0 + dz;
          float w = (dx ? fx : 1.0f - fx) * (dy ? fy : 1.0f - fy) * (dz ? fz : 1.0f - fz);
          if ((xi >= 0) && (xi < W_) && (yi >= 0) && (yi < H_) && (zi >= 0) && (zi < DBINS)) {
            agg  += w * dpn[((size_t)cam * DBINS + zi) * HW + yi * W_ + xi];
            msum += w;
          }
        }
  }
  out[i] = (msum > 0.0f) ? (agg / msum) : agg;
}

extern "C" void kernel_launch(void* const* d_in, const int* in_sizes, int n_in,
                              void* d_out, int out_size, void* d_ws, size_t ws_size,
                              hipStream_t stream) {
  (void)in_sizes; (void)n_in; (void)out_size; (void)ws_size;
  const float* img   = (const float*)d_in[0];
  const float* intr  = (const float*)d_in[1];
  const float* grid  = (const float*)d_in[2];
  const float* red_w = (const float*)d_in[3];
  const float* red_s = (const float*)d_in[4];
  const float* red_b = (const float*)d_in[5];
  const float* w1    = (const float*)d_in[6];
  const float* b1    = (const float*)d_in[7];
  const float* w2    = (const float*)d_in[8];
  const float* b2    = (const float*)d_in[9];
  const float* se_rw = (const float*)d_in[10];
  const float* se_rb = (const float*)d_in[11];
  const float* se_ew = (const float*)d_in[12];
  const float* se_eb = (const float*)d_in[13];
  const float* bb_w  = (const float*)d_in[14];
  const float* bb_s  = (const float*)d_in[15];
  const float* bb_b  = (const float*)d_in[16];
  const float* dp_w  = (const float*)d_in[17];
  const float* dp_b  = (const float*)d_in[18];
  float* out = (float*)d_out;

  char* ws = (char*)d_ws;
  size_t off = 0;
  auto alloc = [&](size_t bytes) -> char* {
    char* p = ws + off;
    off = (off + bytes + 255) & ~(size_t)255;
    return p;
  };
  _Float16* imgT = (_Float16*)alloc((size_t)NCAM * HW * CIN * 2);
  _Float16* wtR  = (_Float16*)alloc((size_t)MID * 9 * CIN * 2);
  _Float16* wtB  = (_Float16*)alloc((size_t)6 * MID * 9 * MID * 2);
  _Float16* wtD  = (_Float16*)alloc((size_t)DBINS * MID * 2);
  _Float16* X    = (_Float16*)alloc((size_t)NPIX * MID * 2);
  _Float16* Y    = (_Float16*)alloc((size_t)NPIX * MID * 2);
  float*    dep  = (float*)alloc((size_t)NPIX * DBINS * 4);
  float*    dpn  = (float*)alloc((size_t)NCAM * DBINS * HW * 4);
  float*    gate = (float*)alloc((size_t)NCAM * MID * 4);

  // layout transforms (f32 -> f16, WMMA-friendly)
  { int t = MID * CIN * 9;     k_wt<<<(t + 255) / 256, 256, 0, stream>>>(red_w, wtR, MID, CIN, 9); }
  { int t = 6 * MID * MID * 9; k_wt<<<(t + 255) / 256, 256, 0, stream>>>(bb_w, wtB, 6 * MID, MID, 9); }
  { int t = DBINS * MID;       k_wt<<<(t + 255) / 256, 256, 0, stream>>>(dp_w, wtD, DBINS, MID, 1); }
  { int t = NCAM * CIN * HW;   k_imgT<<<(t + 255) / 256, 256, 0, stream>>>(img, imgT); }
  k_se<<<NCAM, MID, 0, stream>>>(intr, w1, b1, w2, b2, se_rw, se_rb, se_ew, se_eb, gate);

  dim3 blk(32);
  dim3 gmain(HW / 64, MID / 32, NCAM);  // (44, 8, 6)
  // reduce conv 512->256, fused scale/bias/relu/SE-gate
  k_conv<CIN, 9, 0><<<gmain, blk, 0, stream>>>(imgT, wtR, red_s, red_b, gate, nullptr, X);
  // 3 residual blocks of two 3x3 convs
  for (int i = 0; i < 3; ++i) {
    const _Float16* wa = wtB + (size_t)(i * 2 + 0) * MID * 9 * MID;
    const _Float16* wb = wtB + (size_t)(i * 2 + 1) * MID * 9 * MID;
    k_conv<MID, 9, 1><<<gmain, blk, 0, stream>>>(
        X, wa, bb_s + (i * 2 + 0) * MID, bb_b + (i * 2 + 0) * MID, nullptr, nullptr, Y);
    k_conv<MID, 9, 2><<<gmain, blk, 0, stream>>>(
        Y, wb, bb_s + (i * 2 + 1) * MID, bb_b + (i * 2 + 1) * MID, nullptr, X, X);
  }
  // depth head 1x1 conv 256->112 (f32 out)
  k_head<<<dim3(HW / 32, DBINS / 16, NCAM), blk, 0, stream>>>(X, wtD, dp_b, dep);
  // softmax + 3D trilinear sample / camera-sum / normalize
  k_softmax<<<(NPIX + 255) / 256, 256, 0, stream>>>(dep, dpn);
  k_sample<<<(VXD * VYD * VZD + 255) / 256, 256, 0, stream>>>(grid, dpn, out);
}
